// MultiScalePyramidAttention_85985245265955
// MI455X (gfx1250) — compile-verified, matrix operand order checked by reading, not run on hardware
//
#include <hip/hip_runtime.h>
#include <hip/hip_bf16.h>
#include <math.h>

typedef __bf16 bf16_t;
typedef __attribute__((ext_vector_type(16))) __bf16 v16bf;
typedef __attribute__((ext_vector_type(8)))  float   v8f;
typedef __attribute__((ext_vector_type(4)))  int     v4i;

typedef __attribute__((address_space(1))) v4i* gptr_v4i;
typedef __attribute__((address_space(3))) v4i* lptr_v4i;

#define BATCH 8
#define C 128
#define H 64
#define NPIX (BATCH * H * H)   // 32768

#if __has_builtin(__builtin_amdgcn_global_load_async_to_lds_b128)
#define HAVE_ASYNC_LDS 1
#endif

// ---------------- WMMA fragment helpers (16x16x32 bf16, wave32) --------------
// 16-bit A (16x32) layout: lane l: m = l&15, g = l>>4.
//   VGPR v (pair e=2v,2v+1): v<4 -> K = g*8 + 2v ; v>=4 -> K = 16 + g*8 + 2(v-4)
// B (32x16) mirrors with lane = N. C/D f32: lane n = l&15, VGPR v -> m = v + 8*(l>>4).
__device__ __forceinline__ int wmma_kbase(int v, int g) {
  return (v < 4) ? (g * 8 + 2 * v) : (16 + g * 8 + 2 * (v - 4));
}

// Load a fragment whose K-dimension is contiguous in memory starting at rowptr.
__device__ __forceinline__ v16bf frag_k_contig(const bf16_t* rowptr, int g) {
  v16bf f;
#pragma unroll
  for (int v = 0; v < 8; ++v) {
    int k = wmma_kbase(v, g);
    f[2 * v]     = rowptr[k];
    f[2 * v + 1] = rowptr[k + 1];
  }
  return f;
}

__device__ __forceinline__ v8f wmma_bf16(v16bf a, v16bf b, v8f c) {
  return __builtin_amdgcn_wmma_f32_16x16x32_bf16(false, a, false, b,
                                                 (short)0, c, false, false);
}

// Async memory->LDS copy of 16 bytes (GLOBAL_LOAD_ASYNC_TO_LDS_B128, ASYNCcnt).
__device__ __forceinline__ void async_copy_b128(const void* g, void* l) {
#ifdef HAVE_ASYNC_LDS
  void* gnc = const_cast<void*>(g);
  __builtin_amdgcn_global_load_async_to_lds_b128(
      (gptr_v4i)gnc, (lptr_v4i)l, 0, 0);
#endif
}

__device__ __forceinline__ void wait_async_zero() {
#if __has_builtin(__builtin_amdgcn_s_wait_asynccnt)
  __builtin_amdgcn_s_wait_asynccnt(0);
#else
  asm volatile("s_wait_asynccnt 0" ::: "memory");
#endif
}

// ---------------- K0: convert weights to bf16 --------------------------------
// conv weights transposed to [scale][r=ky*3+kx][o][c] so K(=c) pairs contiguous.
#define SZ_CONVW (4 * 9 * 128 * 128)   // 589824
#define SZ_AIN   (384 * 128)           // 49152
#define SZ_AOUT  (128 * 128)           // 16384
#define SZ_FUS   (128 * 512)           // 65536
#define SZ_ALL   (SZ_CONVW + SZ_AIN + SZ_AOUT + SZ_FUS)

__global__ void convert_weights_kernel(const float* __restrict__ conv_w,
                                       const float* __restrict__ attn_in_w,
                                       const float* __restrict__ attn_out_w,
                                       const float* __restrict__ fus_w,
                                       bf16_t* __restrict__ wbT_conv,
                                       bf16_t* __restrict__ wb_ain,
                                       bf16_t* __restrict__ wb_aout,
                                       bf16_t* __restrict__ wb_fus) {
  int idx = blockIdx.x * blockDim.x + threadIdx.x;
  if (idx < SZ_CONVW) {
    int c = idx % 128;  int t = idx / 128;
    int o = t % 128;    t /= 128;
    int r = t % 9;      int i = t / 9;
    wbT_conv[idx] = (bf16_t)conv_w[((i * 128 + o) * 128 + c) * 9 + r];
    return;
  }
  idx -= SZ_CONVW;
  if (idx < SZ_AIN)  { wb_ain[idx]  = (bf16_t)attn_in_w[idx];  return; }
  idx -= SZ_AIN;
  if (idx < SZ_AOUT) { wb_aout[idx] = (bf16_t)attn_out_w[idx]; return; }
  idx -= SZ_AOUT;
  if (idx < SZ_FUS)  { wb_fus[idx]  = (bf16_t)fus_w[idx]; }
}

// ---------------- K1: average pooling ---------------------------------------
__global__ void pool_kernel(const float* __restrict__ x, float* __restrict__ out,
                            int factor) {
  int hout = H / factor;
  int total = BATCH * C * hout * hout;
  int idx = blockIdx.x * blockDim.x + threadIdx.x;
  if (idx >= total) return;
  int xo = idx % hout; int t = idx / hout;
  int yo = t % hout;   t /= hout;
  int c  = t % C;      int b = t / C;
  const float* p = x + (((size_t)b * C + c) * H + yo * factor) * H + xo * factor;
  float s = 0.f;
  for (int dy = 0; dy < factor; ++dy)
    for (int dx = 0; dx < factor; ++dx) s += p[dy * H + dx];
  out[idx] = s / (float)(factor * factor);
}

// ---------------- K2: conv3x3 as WMMA implicit GEMM --------------------------
// grid (h/4, h/4, BATCH), block 256 (8 waves). Each wave: 16 out-channels x 16px.
__global__ __launch_bounds__(256)
void conv3x3_kernel(const float* __restrict__ in,     // [B][C][h][h]
                    const bf16_t* __restrict__ wT,    // [9][128][128]
                    const float* __restrict__ bias,   // [128]
                    float* __restrict__ out,          // [B][C][h][h]
                    int h) {
  __shared__ bf16_t patch[6][6][C];   // zero-padded 4x4 tile halo, 9216 B
  int bx = blockIdx.x, by = blockIdx.y, b = blockIdx.z;
  int tid = threadIdx.x;
  for (int e = tid; e < 6 * 6 * C; e += 256) {
    int c = e % C;  int t = e / C;
    int px = t % 6; int py = t / 6;
    int y = by * 4 + py - 1, x = bx * 4 + px - 1;
    float v = 0.f;
    if (y >= 0 && y < h && x >= 0 && x < h)
      v = in[(((size_t)b * C + c) * h + y) * h + x];
    patch[py][px][c] = (bf16_t)v;
  }
  __syncthreads();

  int wave = tid >> 5, lane = tid & 31;
  int g = lane >> 4, nl = lane & 15;
  int py = nl >> 2, px = nl & 3;                 // pixel within 4x4 tile
  v8f acc = {};
#pragma unroll
  for (int r = 0; r < 9; ++r) {
    int ky = r / 3, kx = r % 3;
    const bf16_t* arow = wT + ((size_t)r * 128 + wave * 16 + nl) * 128;
    const bf16_t* brow = &patch[py + ky][px + kx][0];
#pragma unroll
    for (int cc = 0; cc < 4; ++cc) {
      v16bf a = frag_k_contig(arow + cc * 32, g);
      v16bf bm = frag_k_contig(brow + cc * 32, g);
      acc = wmma_bf16(a, bm, acc);
    }
  }
  int y = by * 4 + py, x = bx * 4 + px;
#pragma unroll
  for (int v = 0; v < 8; ++v) {
    int o = wave * 16 + v + 8 * g;
    out[(((size_t)b * C + o) * h + y) * h + x] = acc[v] + bias[o];
  }
}

// ---------------- K3: bilinear upsample + token assembly ---------------------
__global__ void assemble_kernel(const float* __restrict__ f0,
                                const float* __restrict__ f1,
                                const float* __restrict__ f2,
                                const float* __restrict__ f3,
                                bf16_t* __restrict__ tokens) {  // [N][4][128]
  int idx = blockIdx.x * blockDim.x + threadIdx.x;
  if (idx >= NPIX * 4 * C) return;
  int c = idx % C;  int t = idx / C;
  int i = t % 4;    int n = t / 4;
  int x = n % H;    int y = (n / H) % H;  int b = n / (H * H);
  float val;
  if (i == 0) {
    val = f0[(((size_t)b * C + c) * H + y) * H + x];
  } else {
    const float* f = (i == 1) ? f1 : (i == 2) ? f2 : f3;
    int hin = H >> i;                       // 32,16,8
    float sc = (float)hin / (float)H;
    float sy = fmaxf((y + 0.5f) * sc - 0.5f, 0.f);
    float sx = fmaxf((x + 0.5f) * sc - 0.5f, 0.f);
    int y0 = (int)sy; int y1 = (y0 + 1 < hin) ? y0 + 1 : hin - 1; float wy = sy - y0;
    int x0 = (int)sx; int x1 = (x0 + 1 < hin) ? x0 + 1 : hin - 1; float wx = sx - x0;
    const float* base = f + ((size_t)b * C + c) * hin * hin;
    float tl = base[y0 * hin + x0], tr = base[y0 * hin + x1];
    float bl = base[y1 * hin + x0], br = base[y1 * hin + x1];
    val = (tl * (1.f - wy) + bl * wy) * (1.f - wx) +
          (tr * (1.f - wy) + br * wy) * wx;
  }
  tokens[idx] = (bf16_t)val;
}

// ---------------- K4: per-pixel 4-token attention + LN -----------------------
// 64 threads (2 waves), 8 pixels (32 tokens) per block.
__global__ __launch_bounds__(64)
void attention_kernel(const bf16_t* __restrict__ tokens,  // [N][4][128]
                      const bf16_t* __restrict__ w_in,    // [384][128]
                      const float*  __restrict__ b_in,    // [384]
                      const bf16_t* __restrict__ w_out,   // [128][128]
                      const float*  __restrict__ b_out,   // [128]
                      const float*  __restrict__ ln_g,
                      const float*  __restrict__ ln_b,
                      bf16_t* __restrict__ normed) {      // [N][512]
  __shared__ float  lds_qkv[32][384];   // 49152 B; reused as res[32][128]
  __shared__ bf16_t lds_o[32][128];     // 8192 B
  int n0 = blockIdx.x * 8;              // first pixel
  int tid = threadIdx.x;
  int wave = tid >> 5, lane = tid & 31;
  int g = lane >> 4, nl = lane & 15;

  // ---- QKV projection: M=32 tokens (16/wave), N=384, K=128 ----
  int tokm = wave * 16 + nl;            // A-row (token) for this lane
  const bf16_t* arow =
      tokens + ((size_t)(n0 + (tokm >> 2)) * 4 + (tokm & 3)) * 128;
  v16bf afr[4];
#pragma unroll
  for (int cc = 0; cc < 4; ++cc) afr[cc] = frag_k_contig(arow + cc * 32, g);

  for (int t = 0; t < 24; ++t) {
    v8f acc = {};
    const bf16_t* brow = w_in + (size_t)(t * 16 + nl) * 128;
#pragma unroll
    for (int cc = 0; cc < 4; ++cc)
      acc = wmma_bf16(afr[cc], frag_k_contig(brow + cc * 32, g), acc);
    float bias = b_in[t * 16 + nl];
#pragma unroll
    for (int v = 0; v < 8; ++v) {
      int m = wave * 16 + v + 8 * g;
      lds_qkv[m][t * 16 + nl] = acc[v] + bias;
    }
  }
  __syncthreads();

  // ---- attention: one (pixel, head) per thread, fp32 ----
  {
    int pix = tid >> 3, head = tid & 7;
    int base = pix * 4, hoff = head * 16;
    float sc[4][4];
#pragma unroll
    for (int i = 0; i < 4; ++i)
#pragma unroll
      for (int j = 0; j < 4; ++j) {
        float s = 0.f;
        for (int d = 0; d < 16; ++d)
          s += lds_qkv[base + i][hoff + d] * lds_qkv[base + j][128 + hoff + d];
        sc[i][j] = s * 0.25f;   // 1/sqrt(16)
      }
#pragma unroll
    for (int i = 0; i < 4; ++i) {
      float m = fmaxf(fmaxf(sc[i][0], sc[i][1]), fmaxf(sc[i][2], sc[i][3]));
      float e0 = __expf(sc[i][0] - m), e1 = __expf(sc[i][1] - m);
      float e2 = __expf(sc[i][2] - m), e3 = __expf(sc[i][3] - m);
      float inv = 1.f / (e0 + e1 + e2 + e3);
      for (int d = 0; d < 16; ++d) {
        float o = (e0 * lds_qkv[base + 0][256 + hoff + d] +
                   e1 * lds_qkv[base + 1][256 + hoff + d] +
                   e2 * lds_qkv[base + 2][256 + hoff + d] +
                   e3 * lds_qkv[base + 3][256 + hoff + d]) * inv;
        lds_o[base + i][hoff + d] = (bf16_t)o;
      }
    }
  }
  __syncthreads();

  // ---- out-projection + residual: M=32 tokens, N=128, K=128 ----
  v16bf ofr[4];
#pragma unroll
  for (int cc = 0; cc < 4; ++cc)
    ofr[cc] = frag_k_contig(&lds_o[wave * 16 + nl][cc * 32], g);

  for (int t = 0; t < 8; ++t) {
    v8f acc = {};
    const bf16_t* brow = w_out + (size_t)(t * 16 + nl) * 128;
#pragma unroll
    for (int cc = 0; cc < 4; ++cc)
      acc = wmma_bf16(ofr[cc], frag_k_contig(brow + cc * 32, g), acc);
    float bias = b_out[t * 16 + nl];
#pragma unroll
    for (int v = 0; v < 8; ++v) {
      int m = wave * 16 + v + 8 * g;
      float res = acc[v] + bias +
          (float)tokens[((size_t)(n0 + (m >> 2)) * 4 + (m & 3)) * 128 + t * 16 + nl];
      lds_qkv[m][t * 16 + nl] = res;   // reuse qkv LDS as residual buffer
    }
  }
  __syncthreads();

  // ---- LayerNorm per token, emit bf16 normed tokens ----
  if (tid < 32) {
    float mu = 0.f;
    for (int c = 0; c < C; ++c) mu += lds_qkv[tid][c];
    mu *= (1.f / C);
    float var = 0.f;
    for (int c = 0; c < C; ++c) {
      float d = lds_qkv[tid][c] - mu;
      var += d * d;
    }
    var *= (1.f / C);
    float inv = rsqrtf(var + 1e-5f);
    bf16_t* dst = normed + ((size_t)(n0 + (tid >> 2)) * 4 + (tid & 3)) * 128;
    for (int c = 0; c < C; ++c)
      dst[c] = (bf16_t)((lds_qkv[tid][c] - mu) * inv * ln_g[c] + ln_b[c]);
  }
}

// ---------------- K5: fused 1x1 GEMM: M=128, K=512, 16 px / block ------------
// Activation tile staged memory->LDS with GLOBAL_LOAD_ASYNC_TO_LDS_B128
// (ASYNCcnt + s_wait_asynccnt) when available — no VGPR round-trip.
__global__ __launch_bounds__(256)
void fused_kernel(const bf16_t* __restrict__ normed,  // [N][512]
                  const bf16_t* __restrict__ wfus,    // [128][512]
                  const float*  __restrict__ fbias,   // [128]
                  float* __restrict__ out) {          // [B][128][64][64]
  __shared__ bf16_t btile[16][512];                   // 16 KB
  int n0 = blockIdx.x * 16;
  int tid = threadIdx.x;
#ifdef HAVE_ASYNC_LDS
  {
    const char* gsrc = (const char*)(normed + (size_t)n0 * 512);
    char* lbase = (char*)&btile[0][0];
#pragma unroll
    for (int pass = 0; pass < 4; ++pass) {
      int off = pass * 4096 + tid * 16;
      async_copy_b128(gsrc + off, lbase + off);
    }
    wait_async_zero();
  }
#else
  {
    const unsigned int* src = (const unsigned int*)(normed + (size_t)n0 * 512);
    unsigned int* dstu = (unsigned int*)&btile[0][0];
    for (int e = tid; e < 4096; e += 256) dstu[e] = src[e];
  }
#endif
  __syncthreads();

  int wave = tid >> 5, lane = tid & 31;
  int g = lane >> 4, nl = lane & 15;
  const bf16_t* arow = wfus + (size_t)(wave * 16 + nl) * 512;
  v8f acc = {};
#pragma unroll
  for (int cc = 0; cc < 16; ++cc) {
    if (cc + 1 < 16) __builtin_prefetch(arow + (cc + 1) * 32, 0, 3);
    v16bf a = frag_k_contig(arow + cc * 32, g);
    v16bf bm = frag_k_contig(&btile[nl][cc * 32], g);
    acc = wmma_bf16(a, bm, acc);
  }
  int n = n0 + nl;
  int b = n >> 12, yx = n & 4095;
#pragma unroll
  for (int v = 0; v < 8; ++v) {
    int o = wave * 16 + v + 8 * g;
    __builtin_nontemporal_store(acc[v] + fbias[o],
                                &out[((size_t)b * C + o) * 4096 + yx]);
  }
}

// ---------------- launch -----------------------------------------------------
extern "C" void kernel_launch(void* const* d_in, const int* in_sizes, int n_in,
                              void* d_out, int out_size, void* d_ws, size_t ws_size,
                              hipStream_t stream) {
  const float* x          = (const float*)d_in[0];
  const float* conv_w     = (const float*)d_in[1];
  const float* conv_b     = (const float*)d_in[2];
  const float* attn_in_w  = (const float*)d_in[3];
  const float* attn_in_b  = (const float*)d_in[4];
  const float* attn_out_w = (const float*)d_in[5];
  const float* attn_out_b = (const float*)d_in[6];
  const float* ln_g       = (const float*)d_in[7];
  const float* ln_b       = (const float*)d_in[8];
  const float* fus_w      = (const float*)d_in[9];
  const float* fus_b      = (const float*)d_in[10];
  float* out = (float*)d_out;

  char* ws = (char*)d_ws;
  bf16_t* wbT_conv = (bf16_t*)(ws + 0);                 // 1,179,648 B
  bf16_t* wb_ain   = (bf16_t*)(ws + 1179648);           //    98,304 B
  bf16_t* wb_aout  = (bf16_t*)(ws + 1277952);           //    32,768 B
  bf16_t* wb_fus   = (bf16_t*)(ws + 1310720);           //   131,072 B
  float*  x2       = (float*) (ws + 1441792);           // 4,194,304 B
  float*  x4       = (float*) (ws + 5636096);           // 1,048,576 B
  float*  x8       = (float*) (ws + 6684672);           //   262,144 B
  float*  f0       = (float*) (ws + 6946816);           //16,777,216 B
  float*  f1       = (float*) (ws + 23724032);          // 4,194,304 B
  float*  f2       = (float*) (ws + 27918336);          // 1,048,576 B
  float*  f3       = (float*) (ws + 28966912);          //   262,144 B
  bf16_t* tokens   = (bf16_t*)(ws + 29229056);          //33,554,432 B
  bf16_t* normed   = (bf16_t*)(ws + 62783488);          //33,554,432 B  (total ~92 MB)

  // K0: weight conversion
  convert_weights_kernel<<<(SZ_ALL + 255) / 256, 256, 0, stream>>>(
      conv_w, attn_in_w, attn_out_w, fus_w, wbT_conv, wb_ain, wb_aout, wb_fus);

  // K1: pools
  pool_kernel<<<(BATCH * C * 32 * 32 + 255) / 256, 256, 0, stream>>>(x, x2, 2);
  pool_kernel<<<(BATCH * C * 16 * 16 + 255) / 256, 256, 0, stream>>>(x, x4, 4);
  pool_kernel<<<(BATCH * C * 8 * 8 + 255) / 256, 256, 0, stream>>>(x, x8, 8);

  // K2: conv3x3 per scale
  conv3x3_kernel<<<dim3(16, 16, BATCH), 256, 0, stream>>>(
      x,  wbT_conv + 0 * 9 * 128 * 128, conv_b + 0 * C, f0, 64);
  conv3x3_kernel<<<dim3(8, 8, BATCH), 256, 0, stream>>>(
      x2, wbT_conv + 1 * 9 * 128 * 128, conv_b + 1 * C, f1, 32);
  conv3x3_kernel<<<dim3(4, 4, BATCH), 256, 0, stream>>>(
      x4, wbT_conv + 2 * 9 * 128 * 128, conv_b + 2 * C, f2, 16);
  conv3x3_kernel<<<dim3(2, 2, BATCH), 256, 0, stream>>>(
      x8, wbT_conv + 3 * 9 * 128 * 128, conv_b + 3 * C, f3, 8);

  // K3: upsample + token assembly
  assemble_kernel<<<(NPIX * 4 * C + 255) / 256, 256, 0, stream>>>(
      f0, f1, f2, f3, tokens);

  // K4: attention + LN (8 pixels / block)
  attention_kernel<<<NPIX / 8, 64, 0, stream>>>(
      tokens, wb_ain, attn_in_b, wb_aout, attn_out_b, ln_g, ln_b, normed);

  // K5: fused 1x1 projection (16 pixels / block)
  fused_kernel<<<NPIX / 16, 256, 0, stream>>>(normed, wb_fus, fus_b, out);
}
